// RBF_23922967839461
// MI455X (gfx1250) — matrix operand order repaired and checked by reading, not compiled, not run on hardware
//
#include <hip/hip_runtime.h>
#include <hip/hip_bf16.h>

#define B_ROWS  8192
#define IN_DIM  512
#define OUT_DIM 1024

#define BM 128
#define BN 128
#define BK 32
#define LDSTRIDE 40   // padded LDS row stride in bf16 (32 data + 8 pad) = 80 B, 16B-aligned

typedef __bf16 bf16;
typedef __attribute__((ext_vector_type(4)))  __bf16 v4bf;
typedef __attribute__((ext_vector_type(16))) __bf16 v16bf;
typedef __attribute__((ext_vector_type(8)))  float  v8f;

static __device__ __forceinline__ bf16 f2bf(float f) {
  union { float f; unsigned u; } x; x.f = f;
  unsigned r = x.u + 0x7FFFu + ((x.u >> 16) & 1u);   // round-to-nearest-even
  unsigned short h = (unsigned short)(r >> 16);
  return __builtin_bit_cast(bf16, h);
}

// Async DMA: copy 16 bytes/lane from global to LDS, tracked by ASYNCcnt.
static __device__ __forceinline__ void async_copy16(const bf16* g, bf16* l) {
  unsigned loff = (unsigned)(unsigned long long)(__attribute__((address_space(3))) bf16*)l;
  unsigned long long ga = (unsigned long long)g;
  asm volatile("global_load_async_to_lds_b128 %0, %1, off"
               :: "v"(loff), "v"(ga) : "memory");
}

// ---------------------------------------------------------------------------
// Prep: bf16 mirrors of X and C in workspace + row squared-norms + exp(-2 ls).
// One wave32 per row (8192 X rows, then 1024 C rows).
// ---------------------------------------------------------------------------
__global__ __launch_bounds__(256)
void rbf_prep(const float* __restrict__ X, const float* __restrict__ C,
              const float* __restrict__ logs,
              bf16* __restrict__ Xb, bf16* __restrict__ Cb,
              float* __restrict__ xsq, float* __restrict__ csq,
              float* __restrict__ inv2) {
  const int row  = blockIdx.x * 8 + (threadIdx.x >> 5);
  const int lane = threadIdx.x & 31;
  const bool isX = row < B_ROWS;
  const float* src = isX ? (X + (size_t)row * IN_DIM)
                         : (C + (size_t)(row - B_ROWS) * IN_DIM);
  bf16* dst = isX ? (Xb + (size_t)row * IN_DIM)
                  : (Cb + (size_t)(row - B_ROWS) * IN_DIM);
  const float4* p = (const float4*)src;
  float s = 0.0f;
#pragma unroll
  for (int i = 0; i < 4; ++i) {
    const int j = lane + i * 32;
    float4 v = p[j];
    s += v.x * v.x + v.y * v.y + v.z * v.z + v.w * v.w;
    *(v4bf*)(dst + j * 4) = (v4bf){f2bf(v.x), f2bf(v.y), f2bf(v.z), f2bf(v.w)};
  }
#pragma unroll
  for (int off = 16; off; off >>= 1) s += __shfl_xor(s, off, 32);
  if (lane == 0) {
    if (isX) {
      xsq[row] = s;
    } else {
      const int o = row - B_ROWS;
      csq[o]  = s;
      inv2[o] = __expf(-2.0f * logs[o]);
    }
  }
}

// Plain row-norm kernel for the no-workspace fallback path.
__global__ __launch_bounds__(256)
void rbf_rowsq(const float* __restrict__ X, const float* __restrict__ C,
               const float* __restrict__ logs,
               float* __restrict__ xsq, float* __restrict__ csq,
               float* __restrict__ inv2) {
  const int row  = blockIdx.x * 8 + (threadIdx.x >> 5);
  const int lane = threadIdx.x & 31;
  const float* base = (row < B_ROWS) ? (X + (size_t)row * IN_DIM)
                                     : (C + (size_t)(row - B_ROWS) * IN_DIM);
  const float4* p = (const float4*)base;
  float s = 0.0f;
#pragma unroll
  for (int i = 0; i < 4; ++i) {
    float4 v = p[lane + i * 32];
    s += v.x * v.x + v.y * v.y + v.z * v.z + v.w * v.w;
  }
#pragma unroll
  for (int off = 16; off; off >>= 1) s += __shfl_xor(s, off, 32);
  if (lane == 0) {
    if (row < B_ROWS) {
      xsq[row] = s;
    } else {
      const int o = row - B_ROWS;
      csq[o]  = s;
      inv2[o] = __expf(-2.0f * logs[o]);
    }
  }
}

// ---- fragment builders per CDNA5 16-bit VGPR layouts (ISA 7.12.2) ----------
static __device__ __forceinline__ v16bf load_frag_a(const bf16* base, int g) {
  // A 16x32: lane m=M; half-wave g=0 K={0..7,16..23}, g=1 K={8..15,24..31}
  const bf16* b0 = base + 8 * g;
  v4bf c0 = *(const v4bf*)(b0);
  v4bf c1 = *(const v4bf*)(b0 + 4);
  v4bf c2 = *(const v4bf*)(b0 + 16);
  v4bf c3 = *(const v4bf*)(b0 + 20);
  v16bf a;
#pragma unroll
  for (int i = 0; i < 4; ++i) { a[i]=c0[i]; a[4+i]=c1[i]; a[8+i]=c2[i]; a[12+i]=c3[i]; }
  return a;
}
static __device__ __forceinline__ v16bf load_frag_b(const bf16* base, int g) {
  // B 32x16: lane n=N; half-wave g=0 K={0..7,16..23}, g=1 K={8..15,24..31}
  const bf16* b0 = base + 8 * g;
  v4bf c0 = *(const v4bf*)(b0);
  v4bf c1 = *(const v4bf*)(b0 + 4);
  v4bf c2 = *(const v4bf*)(b0 + 16);
  v4bf c3 = *(const v4bf*)(b0 + 20);
  v16bf b;
#pragma unroll
  for (int i = 0; i < 4; ++i) { b[i]=c0[i]; b[4+i]=c1[i]; b[8+i]=c2[i]; b[12+i]=c3[i]; }
  return b;
}

// ---------------------------------------------------------------------------
// Main kernel (async path): bf16 operands, double-buffered LDS tiles fed by
// GLOBAL_LOAD_ASYNC_TO_LDS_B128, 8 waves in a 4(M)x2(N) grid, 2x4 WMMA accs.
// ---------------------------------------------------------------------------
__global__ __launch_bounds__(256, 2)
void rbf_wmma_async(const bf16* __restrict__ Xb, const bf16* __restrict__ Cb,
                    const float* __restrict__ xsq, const float* __restrict__ csq,
                    const float* __restrict__ inv2, float* __restrict__ out) {
  __shared__ bf16 As[2][BM * LDSTRIDE];
  __shared__ bf16 Bs[2][BN * LDSTRIDE];

  const int tid   = threadIdx.x;
  const int lane  = tid & 31;
  const int wid   = tid >> 5;
  const int waveM = wid >> 1;      // 0..3, 32 rows each
  const int waveN = wid & 1;       // 0..1, 64 cols each
  const int g     = lane >> 4;
  const int m16   = lane & 15;

  const int blockM = blockIdx.x * BM;
  const int blockN = blockIdx.y * BN;

  v8f acc[2][4] = {};

  // Stage tile kt into LDS buffer `buf`: 4 async ops per thread/wave.
  auto stage = [&](int kt, int buf) {
    const int k0 = kt * BK;
#pragma unroll
    for (int i = 0; i < 2; ++i) {
      const int v  = tid + i * 256;      // 0..511 over a 128x4 grid of 16B chunks
      const int r  = v >> 2;             // tile row 0..127
      const int c8 = (v & 3) << 3;       // bf16 col 0,8,16,24
      async_copy16(Xb + (size_t)(blockM + r) * IN_DIM + k0 + c8,
                   &As[buf][r * LDSTRIDE + c8]);
      async_copy16(Cb + (size_t)(blockN + r) * IN_DIM + k0 + c8,
                   &Bs[buf][r * LDSTRIDE + c8]);
    }
  };

  constexpr int NK = IN_DIM / BK;    // 16
  stage(0, 0);

  for (int kt = 0; kt < NK; ++kt) {
    const int buf = kt & 1;
    if (kt + 1 < NK) {
      stage(kt + 1, buf ^ 1);
      asm volatile("s_wait_asynccnt 0x4" ::: "memory");  // tile kt landed (per wave)
    } else {
      asm volatile("s_wait_asynccnt 0x0" ::: "memory");
    }
    __syncthreads();                                     // all waves' tile kt landed

    v16bf af[2], bfr[4];
#pragma unroll
    for (int mt = 0; mt < 2; ++mt)
      af[mt] = load_frag_a(&As[buf][(waveM * 32 + mt * 16 + m16) * LDSTRIDE], g);
#pragma unroll
    for (int nt = 0; nt < 4; ++nt)
      bfr[nt] = load_frag_b(&Bs[buf][(waveN * 64 + nt * 16 + m16) * LDSTRIDE], g);

#pragma unroll
    for (int mt = 0; mt < 2; ++mt)
#pragma unroll
      for (int nt = 0; nt < 4; ++nt)
        acc[mt][nt] = __builtin_amdgcn_wmma_f32_16x16x32_bf16(
            false, af[mt], false, bfr[nt], (short)0, acc[mt][nt], false, false);

    __syncthreads();   // protect buf from being overwritten by stage(kt+2)
  }

  // Fused epilogue: out = exp(-max(x^2 - 2xc + c^2, 0) * e^{-2 log s})
#pragma unroll
  for (int mt = 0; mt < 2; ++mt) {
    float xs[8];
#pragma unroll
    for (int r = 0; r < 8; ++r)
      xs[r] = xsq[blockM + waveM * 32 + mt * 16 + 8 * g + r];
#pragma unroll
    for (int nt = 0; nt < 4; ++nt) {
      const int col = blockN + waveN * 64 + nt * 16 + m16;
      const float cs = csq[col];
      const float is = inv2[col];
#pragma unroll
      for (int r = 0; r < 8; ++r) {
        const int row = blockM + waveM * 32 + mt * 16 + 8 * g + r;
        const float d2  = fmaxf(xs[r] - 2.0f * acc[mt][nt][r] + cs, 0.0f);
        __builtin_nontemporal_store(__expf(-d2 * is),
                                    out + (size_t)row * OUT_DIM + col);
      }
    }
  }
}

// ---------------------------------------------------------------------------
// Fallback kernel (small workspace): in-kernel f32->bf16 staging (round-1).
// ---------------------------------------------------------------------------
__global__ __launch_bounds__(256, 2)
void rbf_wmma_sync(const float* __restrict__ X, const float* __restrict__ C,
                   const float* __restrict__ xsq, const float* __restrict__ csq,
                   const float* __restrict__ inv2, float* __restrict__ out) {
  __shared__ bf16 As[BM * LDSTRIDE];
  __shared__ bf16 Bs[BN * LDSTRIDE];

  const int tid   = threadIdx.x;
  const int lane  = tid & 31;
  const int wid   = tid >> 5;
  const int waveM = wid >> 1;
  const int waveN = wid & 1;
  const int g     = lane >> 4;
  const int m16   = lane & 15;

  const int blockM = blockIdx.x * BM;
  const int blockN = blockIdx.y * BN;

  v8f acc[2][4] = {};

  for (int k0 = 0; k0 < IN_DIM; k0 += BK) {
    __syncthreads();
#pragma unroll
    for (int i = 0; i < 4; ++i) {
      const int v  = tid + i * 256;
      const int r  = v >> 3;
      const int c4 = (v & 7) << 2;
      float4 a = *(const float4*)(X + (size_t)(blockM + r) * IN_DIM + k0 + c4);
      float4 b = *(const float4*)(C + (size_t)(blockN + r) * IN_DIM + k0 + c4);
      *(v4bf*)(As + r * LDSTRIDE + c4) = (v4bf){f2bf(a.x), f2bf(a.y), f2bf(a.z), f2bf(a.w)};
      *(v4bf*)(Bs + r * LDSTRIDE + c4) = (v4bf){f2bf(b.x), f2bf(b.y), f2bf(b.z), f2bf(b.w)};
    }
    __syncthreads();

    v16bf af[2], bfr[4];
#pragma unroll
    for (int mt = 0; mt < 2; ++mt)
      af[mt] = load_frag_a(As + (waveM * 32 + mt * 16 + m16) * LDSTRIDE, g);
#pragma unroll
    for (int nt = 0; nt < 4; ++nt)
      bfr[nt] = load_frag_b(Bs + (waveN * 64 + nt * 16 + m16) * LDSTRIDE, g);

#pragma unroll
    for (int mt = 0; mt < 2; ++mt)
#pragma unroll
      for (int nt = 0; nt < 4; ++nt)
        acc[mt][nt] = __builtin_amdgcn_wmma_f32_16x16x32_bf16(
            false, af[mt], false, bfr[nt], (short)0, acc[mt][nt], false, false);
  }

#pragma unroll
  for (int mt = 0; mt < 2; ++mt) {
    float xs[8];
#pragma unroll
    for (int r = 0; r < 8; ++r)
      xs[r] = xsq[blockM + waveM * 32 + mt * 16 + 8 * g + r];
#pragma unroll
    for (int nt = 0; nt < 4; ++nt) {
      const int col = blockN + waveN * 64 + nt * 16 + m16;
      const float cs = csq[col];
      const float is = inv2[col];
#pragma unroll
      for (int r = 0; r < 8; ++r) {
        const int row = blockM + waveM * 32 + mt * 16 + 8 * g + r;
        const float d2 = fmaxf(xs[r] - 2.0f * acc[mt][nt][r] + cs, 0.0f);
        __builtin_nontemporal_store(__expf(-d2 * is),
                                    out + (size_t)row * OUT_DIM + col);
      }
    }
  }
}

// ---------------------------------------------------------------------------
extern "C" void kernel_launch(void* const* d_in, const int* in_sizes, int n_in,
                              void* d_out, int out_size, void* d_ws, size_t ws_size,
                              hipStream_t stream) {
  const float* input   = (const float*)d_in[0];  // [8192, 512]
  const float* centres = (const float*)d_in[1];  // [1024, 512]
  const float* logs    = (const float*)d_in[2];  // [1024]
  float* out = (float*)d_out;                    // [8192, 1024]

  const size_t xb_elems = (size_t)B_ROWS * IN_DIM;
  const size_t cb_elems = (size_t)OUT_DIM * IN_DIM;
  const size_t need_big = (xb_elems + cb_elems) * sizeof(bf16)
                        + (size_t)(B_ROWS + 2 * OUT_DIM) * sizeof(float);

  dim3 grid(B_ROWS / BM, OUT_DIM / BN);
  const int prep_blocks = (B_ROWS + OUT_DIM) / 8;

  if (ws_size >= need_big) {
    bf16*  Xb   = (bf16*)d_ws;
    bf16*  Cb   = Xb + xb_elems;
    float* xsq  = (float*)(Cb + cb_elems);
    float* csq  = xsq + B_ROWS;
    float* inv2 = csq + OUT_DIM;
    rbf_prep<<<prep_blocks, 256, 0, stream>>>(input, centres, logs,
                                              Xb, Cb, xsq, csq, inv2);
    rbf_wmma_async<<<grid, 256, 0, stream>>>(Xb, Cb, xsq, csq, inv2, out);
  } else {
    float* xsq  = (float*)d_ws;
    float* csq  = xsq + B_ROWS;
    float* inv2 = csq + OUT_DIM;
    rbf_rowsq<<<prep_blocks, 256, 0, stream>>>(input, centres, logs,
                                               xsq, csq, inv2);
    rbf_wmma_sync<<<grid, 256, 0, stream>>>(input, centres, xsq, csq, inv2, out);
  }
}
